// SpatioTemporalGNN_49039936585946
// MI455X (gfx1250) — compile-verified, matrix operand order checked by reading, not run on hardware
//
#include <hip/hip_runtime.h>
#include <hip/hip_bf16.h>

// ---------------- problem constants ----------------
#define B_     8
#define T_     24
#define N_     256
#define F_     16
#define H_     128
#define HEADS_ 4
#define C_     32
#define E_     4096
#define ET_    (E_ + N_)      // 4352 edges incl self loops
#define BT_    (B_ * T_)      // 192
#define HH_    64
#define PRED_  6
#define BN_SC  0.99999500003749f   // 1/sqrt(1+1e-5)

typedef __attribute__((ext_vector_type(16))) __bf16 bf16x16;
typedef __attribute__((ext_vector_type(8)))  float  floatx8;

// ---- CDNA5 WMMA 16-bit A/B fragment loader (16x32 slab, ISA 7.12.2) ----
// Lane (half, lrow) needs two contiguous 8-element K runs of its row:
//   elements 0..7  -> k = half*8 + 0..7
//   elements 8..15 -> k = 16 + half*8 + 0..7
// so the fragment is 4x float4 vector loads + pack to bf16.
__device__ __forceinline__ bf16x16 load_frag(const float* __restrict__ rowk0, int half) {
  const float* p = rowk0 + half * 8;
  const float4 a0 = *(const float4*)(p);
  const float4 a1 = *(const float4*)(p + 4);
  const float4 a2 = *(const float4*)(p + 16);
  const float4 a3 = *(const float4*)(p + 20);
  bf16x16 f;
  f[0]  = (__bf16)a0.x; f[1]  = (__bf16)a0.y; f[2]  = (__bf16)a0.z; f[3]  = (__bf16)a0.w;
  f[4]  = (__bf16)a1.x; f[5]  = (__bf16)a1.y; f[6]  = (__bf16)a1.z; f[7]  = (__bf16)a1.w;
  f[8]  = (__bf16)a2.x; f[9]  = (__bf16)a2.y; f[10] = (__bf16)a2.z; f[11] = (__bf16)a2.w;
  f[12] = (__bf16)a3.x; f[13] = (__bf16)a3.y; f[14] = (__bf16)a3.z; f[15] = (__bf16)a3.w;
  return f;
}

// WMMA GEMM: C[M,N] = act(A[M,K] @ Bm^T + bias + bias2), Bm stored [N,K] row major.
// One wave -> 16x32 C slab (2 accumulators share the A fragment).
// Block = 4 waves = 128 columns. Requires M%16==0, N%128==0, K%32==0.
template <int ACT>
__global__ void k_wmma_gemm(const float* __restrict__ A, const float* __restrict__ Bm,
                            const float* __restrict__ bias, const float* __restrict__ bias2,
                            float* __restrict__ Cmat, int M, int N, int K) {
  const int wave = threadIdx.x >> 5;
  const int lane = threadIdx.x & 31;
  const int half = lane >> 4;
  const int lrow = lane & 15;
  const int tile_n = (blockIdx.x * 4 + wave) * 32;
  const int tile_m = blockIdx.y * 16;
  if (tile_n >= N || tile_m >= M) return;

  const int arow = tile_m + lrow;
  const float* __restrict__ Arow  = A  + (size_t)arow * K;
  const float* __restrict__ Brow0 = Bm + (size_t)(tile_n + lrow) * K;
  const float* __restrict__ Brow1 = Bm + (size_t)(tile_n + 16 + lrow) * K;

  floatx8 acc0 = {}, acc1 = {};
  for (int k0 = 0; k0 < K; k0 += 32) {
    if (k0 + 32 < K) {              // gfx1250 global_prefetch_b8 of next K slab
      __builtin_prefetch(Arow  + k0 + 32, 0, 1);
      __builtin_prefetch(Brow0 + k0 + 32, 0, 1);
      __builtin_prefetch(Brow1 + k0 + 32, 0, 1);
    }
    const bf16x16 af  = load_frag(Arow  + k0, half);
    const bf16x16 bf0 = load_frag(Brow0 + k0, half);
    const bf16x16 bf1 = load_frag(Brow1 + k0, half);
    acc0 = __builtin_amdgcn_wmma_f32_16x16x32_bf16(false, af, false, bf0,
                                                   (short)0, acc0, false, false);
    acc1 = __builtin_amdgcn_wmma_f32_16x16x32_bf16(false, af, false, bf1,
                                                   (short)0, acc1, false, false);
  }

  const int col0 = tile_n + lrow, col1 = tile_n + 16 + lrow;
  float badd0 = 0.f, badd1 = 0.f;
  if (bias)  { badd0 += bias[col0];  badd1 += bias[col1];  }
  if (bias2) { badd0 += bias2[col0]; badd1 += bias2[col1]; }
#pragma unroll
  for (int r = 0; r < 8; ++r) {
    const int row = tile_m + r + half * 8;  // ISA C/D layout: VGPR r -> M = r + 8*half
    float v0 = acc0[r] + badd0;
    float v1 = acc1[r] + badd1;
    if (ACT == 1) { v0 = fmaxf(v0, 0.f); v1 = fmaxf(v1, 0.f); }
    Cmat[(size_t)row * N + col0] = v0;
    Cmat[(size_t)row * N + col1] = v1;
  }
}

// ---------------- helpers ----------------
__device__ __forceinline__ float sigm(float x) { return 1.f / (1.f + __expf(-x)); }

__device__ __forceinline__ void edge_sd(const int* __restrict__ ei, int e, int& s, int& d) {
  if (e < E_) { s = ei[e]; d = ei[E_ + e]; } else { s = d = e - E_; }
}

// 128x128 transpose (GAT W is [K,N]; GEMM wants [N,K])
__global__ void k_transpose_h(const float* __restrict__ W, float* __restrict__ Wt) {
  const int r = blockIdx.x, c = threadIdx.x;
  Wt[r * H_ + c] = W[c * H_ + r];
}

// ---------------- input projection: h = relu(bn(x @ W_in + b_in)) ----------------
__global__ void k_in_proj(const float* __restrict__ x, const float* __restrict__ W,
                          const float* __restrict__ bias, const float* __restrict__ g,
                          const float* __restrict__ bb, float* __restrict__ out) {
  const int r = blockIdx.x, c = threadIdx.x;
  float acc = bias[c];
#pragma unroll
  for (int f = 0; f < F_; ++f) acc += x[r * F_ + f] * W[f * H_ + c];
  acc = acc * (g[c] * BN_SC) + bb[c];
  out[r * H_ + c] = fmaxf(acc, 0.f);
}

// ---------------- GAT: per-node attention coefficients ----------------
__global__ void k_attn_coef(const float* __restrict__ z, const float* __restrict__ a_src,
                            const float* __restrict__ a_dst, float* __restrict__ as_o,
                            float* __restrict__ ad_o) {
  const int r = blockIdx.x, c = threadIdx.x;
  const int h = c >> 5, cc = c & 31;
  const float v = z[r * H_ + c];
  float ps = v * a_src[h * C_ + cc];
  float pd = v * a_dst[h * C_ + cc];
#pragma unroll
  for (int o = 16; o > 0; o >>= 1) { ps += __shfl_down(ps, o); pd += __shfl_down(pd, o); }
  if (cc == 0) { as_o[r * HEADS_ + h] = ps; ad_o[r * HEADS_ + h] = pd; }
}

// ---------------- CSR over dst (built once per launch) ----------------
__global__ void k_csr_count(const int* __restrict__ ei, int* __restrict__ cnt) {
  const int e = blockIdx.x * blockDim.x + threadIdx.x;
  if (e < ET_) { int s, d; edge_sd(ei, e, s, d); atomicAdd(&cnt[d], 1); }
}

__global__ void k_csr_scan(const int* __restrict__ cnt, int* __restrict__ rowptr) {
  __shared__ int sm[N_];
  const int t = threadIdx.x;
  sm[t] = cnt[t];
  __syncthreads();
  for (int off = 1; off < N_; off <<= 1) {
    int v = (t >= off) ? sm[t - off] : 0;
    __syncthreads();
    sm[t] += v;
    __syncthreads();
  }
  rowptr[t + 1] = sm[t];
  if (t == 0) rowptr[0] = 0;
}

__global__ void k_csr_fill(const int* __restrict__ ei, const int* __restrict__ rowptr,
                           int* __restrict__ fill, int* __restrict__ colA) {
  const int e = blockIdx.x * blockDim.x + threadIdx.x;
  if (e < ET_) {
    int s, d; edge_sd(ei, e, s, d);
    const int pos = atomicAdd(&fill[d], 1);
    colA[rowptr[d] + pos] = s;
  }
}

// ---------------- GAT gather: softmax over incoming edges + aggregate + bias + BN --
__global__ void k_gat_gather(const int* __restrict__ rowptr, const int* __restrict__ colA,
                             const float* __restrict__ a_s, const float* __restrict__ a_d,
                             const float* __restrict__ z, const float* __restrict__ bias,
                             const float* __restrict__ g, const float* __restrict__ bb,
                             float* __restrict__ out) {
  const int n = blockIdx.x, bt = blockIdx.y, c = threadIdx.x, h = c >> 5;
  const int beg = rowptr[n], end = rowptr[n + 1];
  const float ad = a_d[(bt * N_ + n) * HEADS_ + h];
  float mxh = -1e30f;
  for (int j = beg; j < end; ++j) {
    float v = a_s[(bt * N_ + colA[j]) * HEADS_ + h] + ad;
    v = (v > 0.f) ? v : 0.2f * v;
    mxh = fmaxf(mxh, v);
  }
  float den = 0.f, acc = 0.f;
  for (int j = beg; j < end; ++j) {
    const int s = colA[j];
    float v = a_s[(bt * N_ + s) * HEADS_ + h] + ad;
    v = (v > 0.f) ? v : 0.2f * v;
    const float p = __expf(v - mxh);
    den += p;
    acc += p * z[(bt * N_ + s) * H_ + c];
  }
  const float o = acc / (den + 1e-16f) + bias[c];
  out[(bt * N_ + n) * H_ + c] = o * (g[c] * BN_SC) + bb[c];
}

// ---------------- LSTM scan (one direction). pre holds Wih*x + bih + bhh ----------
__global__ void k_lstm_scan(const float* __restrict__ pre, const float* __restrict__ Whh,
                            float* __restrict__ out, int out_off, int reverse) {
  const int b = blockIdx.x, tid = threadIdx.x;
  __shared__ float hs[HH_];
  float c = 0.f;
  hs[tid] = 0.f;
  __syncthreads();
  for (int step = 0; step < T_; ++step) {
    const int t = reverse ? (T_ - 1 - step) : step;
    const float* p = pre + (b * T_ + t) * 4 * HH_;
    float gi = p[tid], gf = p[HH_ + tid], gg = p[2 * HH_ + tid], go = p[3 * HH_ + tid];
#pragma unroll 8
    for (int j = 0; j < HH_; ++j) {
      const float hj = hs[j];
      gi += Whh[tid * HH_ + j] * hj;
      gf += Whh[(HH_ + tid) * HH_ + j] * hj;
      gg += Whh[(2 * HH_ + tid) * HH_ + j] * hj;
      go += Whh[(3 * HH_ + tid) * HH_ + j] * hj;
    }
    c = sigm(gf) * c + sigm(gi) * tanhf(gg);
    const float hn = sigm(go) * tanhf(c);
    __syncthreads();
    hs[tid] = hn;
    __syncthreads();
    out[(b * T_ + t) * (2 * HH_) + out_off + tid] = hn;
  }
}

// ---------------- LayerNorm over H=128, optional residual add ----------------
__global__ void k_layernorm(const float* __restrict__ in, const float* __restrict__ resid,
                            const float* __restrict__ g, const float* __restrict__ b,
                            float* __restrict__ out) {
  __shared__ float sm[H_], sq[H_];
  const int r = blockIdx.x, c = threadIdx.x;
  float v = in[r * H_ + c];
  if (resid) v += resid[r * H_ + c];
  sm[c] = v; sq[c] = v * v;
  __syncthreads();
  for (int s = H_ / 2; s > 0; s >>= 1) {
    if (c < s) { sm[c] += sm[c + s]; sq[c] += sq[c + s]; }
    __syncthreads();
  }
  const float mean = sm[0] * (1.f / H_);
  const float var = sq[0] * (1.f / H_) - mean * mean;
  out[r * H_ + c] = (v - mean) * rsqrtf(var + 1e-5f) * g[c] + b[c];
}

// ---------------- MHA core: per (b, head), T=24, d=32 ----------------
__global__ void k_attn(const float* __restrict__ qkv, float* __restrict__ o) {
  const int b = blockIdx.x >> 2, h = blockIdx.x & 3;
  __shared__ float Ks[T_][C_], Vs[T_][C_];
  for (int i = threadIdx.x; i < T_ * C_; i += 32) {
    const int t = i >> 5, d = i & 31;
    Ks[t][d] = qkv[(b * T_ + t) * (3 * H_) + H_ + h * C_ + d];
    Vs[t][d] = qkv[(b * T_ + t) * (3 * H_) + 2 * H_ + h * C_ + d];
  }
  __syncthreads();
  const int q = threadIdx.x;
  if (q < T_) {
    float qr[C_];
#pragma unroll
    for (int d = 0; d < C_; ++d) qr[d] = qkv[(b * T_ + q) * (3 * H_) + h * C_ + d];
    float s[T_];
    float mx = -1e30f;
    for (int t = 0; t < T_; ++t) {
      float acc = 0.f;
#pragma unroll
      for (int d = 0; d < C_; ++d) acc += qr[d] * Ks[t][d];
      acc *= 0.1767766953f;   // 1/sqrt(32)
      s[t] = acc; mx = fmaxf(mx, acc);
    }
    float den = 0.f;
    for (int t = 0; t < T_; ++t) { s[t] = __expf(s[t] - mx); den += s[t]; }
    const float inv = 1.f / den;
#pragma unroll
    for (int d = 0; d < C_; ++d) {
      float acc = 0.f;
      for (int t = 0; t < T_; ++t) acc += s[t] * Vs[t][d];
      o[(b * T_ + q) * H_ + h * C_ + d] = acc * inv;
    }
  }
}

// ---------------- head ----------------
__global__ void k_head1(const float* __restrict__ ln2, const float* __restrict__ W1,
                        const float* __restrict__ b1, float* __restrict__ hid) {
  const int b = blockIdx.x, hh = threadIdx.x;
  float acc = b1[hh];
  const float* last = ln2 + (b * T_ + (T_ - 1)) * H_;
#pragma unroll 8
  for (int k = 0; k < H_; ++k) acc += last[k] * W1[k * HH_ + hh];
  hid[b * HH_ + hh] = fmaxf(acc, 0.f);
}

__global__ void k_head2(const float* __restrict__ hid, const float* __restrict__ W2,
                        const float* __restrict__ b2, float* __restrict__ out) {
  const int b = blockIdx.x;
  const int n = blockIdx.y * blockDim.x + threadIdx.x;   // 0..1535
  float acc = b2[n];
#pragma unroll 8
  for (int k = 0; k < HH_; ++k) acc += hid[b * HH_ + k] * W2[k * (N_ * PRED_) + n];
  out[b * (N_ * PRED_) + n] = acc;
}

// =======================================================================
extern "C" void kernel_launch(void* const* d_in, const int* in_sizes, int n_in,
                              void* d_out, int out_size, void* d_ws, size_t ws_size,
                              hipStream_t stream) {
  // ---- inputs in setup_inputs() insertion order (nested dicts/lists in order) ----
  const float* x  = (const float*)d_in[0];
  const int*   ei = (const int*)d_in[1];
  const float* W_in    = (const float*)d_in[2];
  const float* b_in    = (const float*)d_in[3];
  const float* bn_in_g = (const float*)d_in[4];
  const float* bn_in_b = (const float*)d_in[5];
  struct GatP { const float *W, *a_src, *a_dst, *bias, *bn_g, *bn_b; } gat[2];
  for (int l = 0; l < 2; ++l) {
    const int o = 6 + l * 6;
    gat[l] = { (const float*)d_in[o], (const float*)d_in[o + 1], (const float*)d_in[o + 2],
               (const float*)d_in[o + 3], (const float*)d_in[o + 4], (const float*)d_in[o + 5] };
  }
  struct LstmD { const float *Wih, *Whh, *bih, *bhh; } lstm[2][2];
  for (int l = 0; l < 2; ++l)
    for (int dir = 0; dir < 2; ++dir) {
      const int o = 18 + l * 8 + dir * 4;
      lstm[l][dir] = { (const float*)d_in[o], (const float*)d_in[o + 1],
                       (const float*)d_in[o + 2], (const float*)d_in[o + 3] };
    }
  const float* ln1_g = (const float*)d_in[34];
  const float* ln1_b = (const float*)d_in[35];
  const float* ln2_g = (const float*)d_in[36];
  const float* ln2_b = (const float*)d_in[37];
  const float* Wqkv  = (const float*)d_in[38];
  const float* bqkv  = (const float*)d_in[39];
  const float* Wo    = (const float*)d_in[40];
  const float* bo    = (const float*)d_in[41];
  const float* W1    = (const float*)d_in[42];
  const float* b1    = (const float*)d_in[43];
  const float* W2    = (const float*)d_in[44];
  const float* b2    = (const float*)d_in[45];
  float* out = (float*)d_out;

  // ---- workspace carve (~76 MB) ----
  float* ws = (float*)d_ws;
  size_t off = 0;
  auto alloc = [&](size_t n) { float* p = ws + off; off += n; return p; };
  const size_t BIG = (size_t)BT_ * N_ * H_;     // 6.29M floats
  float* hA   = alloc(BIG);
  float* hB   = alloc(BIG);
  float* zb   = alloc(BIG);
  float* Wt   = alloc((size_t)H_ * H_);
  float* a_s  = alloc((size_t)BT_ * N_ * HEADS_);
  float* a_d  = alloc((size_t)BT_ * N_ * HEADS_);
  float* pre_f = alloc((size_t)BT_ * 4 * HH_);
  float* pre_b = alloc((size_t)BT_ * 4 * HH_);
  float* xl1  = alloc((size_t)BT_ * 2 * HH_);
  float* xl2  = alloc((size_t)BT_ * 2 * HH_);
  float* ln1o = alloc((size_t)BT_ * H_);
  float* qkv  = alloc((size_t)BT_ * 3 * H_);
  float* atto = alloc((size_t)BT_ * H_);
  float* proj = alloc((size_t)BT_ * H_);
  float* ln2o = alloc((size_t)BT_ * H_);
  float* hid  = alloc((size_t)B_ * HH_);
  int* cnt    = (int*)alloc(N_);
  int* fill   = (int*)alloc(N_);
  int* rowptr = (int*)alloc(N_ + 1);
  int* colA   = (int*)alloc(ET_);
  (void)ws_size; (void)in_sizes; (void)n_in; (void)out_size;

  const int ROWS = BT_ * N_;                    // 49152

  // 0) build dst-CSR of the (static) edge list
  hipMemsetAsync(cnt, 0, N_ * sizeof(int), stream);
  hipMemsetAsync(fill, 0, N_ * sizeof(int), stream);
  k_csr_count<<<(ET_ + 255) / 256, 256, 0, stream>>>(ei, cnt);
  k_csr_scan<<<1, N_, 0, stream>>>(cnt, rowptr);
  k_csr_fill<<<(ET_ + 255) / 256, 256, 0, stream>>>(ei, rowptr, fill, colA);

  // 1) input projection
  k_in_proj<<<ROWS, H_, 0, stream>>>(x, W_in, b_in, bn_in_g, bn_in_b, hA);

  // 2) GAT layers
  float* hcur = hA;
  float* hnxt = hB;
  for (int l = 0; l < 2; ++l) {
    k_transpose_h<<<H_, H_, 0, stream>>>(gat[l].W, Wt);
    // z = h @ W  (49152x128 @ 128x128)  -- WMMA, B as [N,K]
    k_wmma_gemm<0><<<dim3(H_ / 128, ROWS / 16), 128, 0, stream>>>(
        hcur, Wt, nullptr, nullptr, zb, ROWS, H_, H_);
    k_attn_coef<<<ROWS, H_, 0, stream>>>(zb, gat[l].a_src, gat[l].a_dst, a_s, a_d);
    k_gat_gather<<<dim3(N_, BT_), H_, 0, stream>>>(rowptr, colA, a_s, a_d, zb,
                                                   gat[l].bias, gat[l].bn_g, gat[l].bn_b,
                                                   hnxt);
    float* t = hcur; hcur = hnxt; hnxt = t;
  }
  // hcur is x_lstm (BT, N*H = 32768)

  // 3) LSTM stack
  const float* xin = hcur;
  int din = N_ * H_;
  float* xout = xl1;
  for (int l = 0; l < 2; ++l) {
    // pre = x @ Wih^T + bih + bhh   -- WMMA (M=192, N=256, K=din), Wih already [N,K]
    k_wmma_gemm<0><<<dim3((4 * HH_) / 128, BT_ / 16), 128, 0, stream>>>(
        xin, lstm[l][0].Wih, lstm[l][0].bih, lstm[l][0].bhh, pre_f, BT_, 4 * HH_, din);
    k_wmma_gemm<0><<<dim3((4 * HH_) / 128, BT_ / 16), 128, 0, stream>>>(
        xin, lstm[l][1].Wih, lstm[l][1].bih, lstm[l][1].bhh, pre_b, BT_, 4 * HH_, din);
    k_lstm_scan<<<B_, HH_, 0, stream>>>(pre_f, lstm[l][0].Whh, xout, 0, 0);
    k_lstm_scan<<<B_, HH_, 0, stream>>>(pre_b, lstm[l][1].Whh, xout, HH_, 1);
    xin = xout;
    din = 2 * HH_;
    xout = xl2;
  }

  // 4) ln1
  k_layernorm<<<BT_, H_, 0, stream>>>(xl2, nullptr, ln1_g, ln1_b, ln1o);

  // 5) MHA
  k_wmma_gemm<0><<<dim3((3 * H_) / 128, BT_ / 16), 128, 0, stream>>>(
      ln1o, Wqkv, bqkv, nullptr, qkv, BT_, 3 * H_, H_);
  k_attn<<<B_ * HEADS_, 32, 0, stream>>>(qkv, atto);
  k_wmma_gemm<0><<<dim3(H_ / 128, BT_ / 16), 128, 0, stream>>>(
      atto, Wo, bo, nullptr, proj, BT_, H_, H_);
  k_layernorm<<<BT_, H_, 0, stream>>>(proj, ln1o, ln2_g, ln2_b, ln2o);

  // 6) head
  k_head1<<<B_, HH_, 0, stream>>>(ln2o, W1, b1, hid);
  k_head2<<<dim3(B_, (N_ * PRED_) / 128), 128, 0, stream>>>(hid, W2, b2, out);
}
